// PCD_31731218383420
// MI455X (gfx1250) — compile-verified, matrix operand order checked by reading, not run on hardware
//
#include <hip/hip_runtime.h>
#include <math.h>

// ---------------------------------------------------------------------------
// Types for CDNA5 WMMA (wave32): v_wmma_f32_16x16x32_bf16
// ---------------------------------------------------------------------------
typedef __attribute__((ext_vector_type(16))) __bf16 v16bf;
typedef __attribute__((ext_vector_type(8)))  __bf16 v8bf;
typedef __attribute__((ext_vector_type(8)))  float  v8f;

#define CH 64          // NF
#define OMCH 216       // 3*DG*K offset/mask channels
#define DCNK 576       // 64*9 reduction depth for DCN gemm

static __device__ __forceinline__ v16bf bf_join(v8bf lo, v8bf hi) {
  v16bf r;
#pragma unroll
  for (int i = 0; i < 8; ++i) { r[i] = lo[i]; r[i + 8] = hi[i]; }
  return r;
}

static __device__ __forceinline__ float lrelu_f(float v) {
  return v >= 0.f ? v : 0.1f * v;
}

// One A-fragment, one B-panel column tile: 1 WMMA per 32-deep K chunk.
static __device__ __forceinline__ v8f wmma_panel(const __bf16* __restrict__ arow,
                                                 const __bf16* brow, int Kp) {
  v8f acc = {};
  for (int kb = 0; kb < Kp; kb += 32) {
    __builtin_prefetch(arow + kb + 256, 0, 0);   // global_prefetch_b8 on weights
    v8bf alo = *(const v8bf*)(arow + kb);
    v8bf ahi = *(const v8bf*)(arow + kb + 16);
    v8bf blo = *(const v8bf*)(brow + kb);
    v8bf bhi = *(const v8bf*)(brow + kb + 16);
    acc = __builtin_amdgcn_wmma_f32_16x16x32_bf16(
        false, bf_join(alo, ahi), false, bf_join(blo, bhi),
        (short)0, acc, false, false);
  }
  return acc;
}

// One A-fragment shared by two B column tiles: 2 WMMAs per A load (halves
// weight traffic, doubles matrix-op density in the hot loop).
static __device__ __forceinline__ void wmma_panel2(const __bf16* __restrict__ arow,
                                                   const __bf16* brow0,
                                                   const __bf16* brow1, int Kp,
                                                   v8f& acc0, v8f& acc1) {
  for (int kb = 0; kb < Kp; kb += 32) {
    __builtin_prefetch(arow + kb + 256, 0, 0);
    v8bf alo = *(const v8bf*)(arow + kb);
    v8bf ahi = *(const v8bf*)(arow + kb + 16);
    v16bf A = bf_join(alo, ahi);
    v8bf b0l = *(const v8bf*)(brow0 + kb);
    v8bf b0h = *(const v8bf*)(brow0 + kb + 16);
    v8bf b1l = *(const v8bf*)(brow1 + kb);
    v8bf b1h = *(const v8bf*)(brow1 + kb + 16);
    acc0 = __builtin_amdgcn_wmma_f32_16x16x32_bf16(
        false, A, false, bf_join(b0l, b0h), (short)0, acc0, false, false);
    acc1 = __builtin_amdgcn_wmma_f32_16x16x32_bf16(
        false, A, false, bf_join(b1l, b1h), (short)0, acc1, false, false);
  }
}

// ---------------------------------------------------------------------------
// Weight pack: f32 OIHW (Cout,Cin,3,3) -> bf16 [CoutPad][Kp], Kp = Cin*9
// ---------------------------------------------------------------------------
__global__ void pack_w_kernel(const float* __restrict__ w, __bf16* __restrict__ out,
                              int Cout, int CoutPad, int Kp) {
  int idx = blockIdx.x * 256 + threadIdx.x;
  int total = CoutPad * Kp;
  if (idx >= total) return;
  int r = idx / Kp;
  int k = idx - r * Kp;
  float v = (r < Cout) ? w[(size_t)r * Kp + k] : 0.f;
  out[idx] = (__bf16)v;
}

// ---------------------------------------------------------------------------
// 3x3 conv (pad 1) as implicit GEMM on WMMA.
//   PXT = pixel tiles (of 16) staged per block: 2 for Kp<=576, 1 for Kp=1152
//   (LDS panel is PXT*16*Kp bf16 = 36 KB either way).
//   Block = 128 threads = 4 waves; wave computes a 16(cout) x PXT*16(pixel)
//   strip: one A-fragment feeds PXT WMMAs per 32-deep K chunk.
// grid = (H*W/(PXT*16), B, ceil(Mtiles/4))
// ---------------------------------------------------------------------------
template <int PXT>
__global__ __launch_bounds__(128)
void conv3x3_wmma(const float* __restrict__ in0, int C0, float s0,
                  const float* __restrict__ in1, int C1, float s1,
                  const __bf16* __restrict__ wpack, const float* __restrict__ bias,
                  float* __restrict__ out, int Cout, int Mtiles,
                  int H, int W, int act) {
  __shared__ __attribute__((aligned(16))) __bf16 ldsB[16 * 1152];
  const int PX = PXT * 16;
  const int Ctot = C0 + C1;
  const int Kp = Ctot * 9;
  const int b = blockIdx.y;
  const int tile = blockIdx.x;
  const int tid = threadIdx.x;

  // ---- stage im2col B-panel: PX pixels x Kp, whole block cooperates
  {
    const int p  = tid % PX;
    const int tc = tid / PX;                 // channel lane
    const int stride = 128 / PX;
    const int pix = tile * PX + p;
    const int y = pix / W;
    const int x = pix - y * W;
    for (int cin = tc; cin < Ctot; cin += stride) {
      const float* src;
      int cc, Csrc; float sc;
      if (cin < C0) { src = in0; cc = cin;      Csrc = C0; sc = s0; }
      else          { src = in1; cc = cin - C0; Csrc = C1; sc = s1; }
      const float* chan = src + ((size_t)b * Csrc + cc) * H * W;
#pragma unroll
      for (int t9 = 0; t9 < 9; ++t9) {
        const int dy = t9 / 3 - 1, dx = t9 % 3 - 1;
        const int yy = y + dy, xx = x + dx;
        float v = 0.f;
        if (yy >= 0 && yy < H && xx >= 0 && xx < W)
          v = chan[yy * W + xx] * sc;
        ldsB[p * Kp + cin * 9 + t9] = (__bf16)v;
      }
    }
  }
  __syncthreads();

  // ---- WMMA strip
  const int wave = tid >> 5;
  const int lane = tid & 31;
  const int mtile = blockIdx.z * 4 + wave;
  if (mtile >= Mtiles) return;
  const int col   = lane & 15;
  const int halfo = (lane >= 16) ? 8 : 0;
  const __bf16* arow = wpack + ((size_t)(mtile * 16 + col)) * Kp + halfo;
  v8f acc[PXT];
  if (PXT == 1) {
    acc[0] = wmma_panel(arow, ldsB + col * Kp + halfo, Kp);
  } else {
    acc[0] = (v8f){};
    acc[PXT - 1] = (v8f){};
    wmma_panel2(arow, ldsB + col * Kp + halfo,
                ldsB + (16 + col) * Kp + halfo, Kp, acc[0], acc[PXT - 1]);
  }

  // ---- epilogue: bias (+ optional lrelu), store f32 NCHW
  const bool full = (mtile * 16 + 16 <= Cout);
#pragma unroll
  for (int u = 0; u < PXT; ++u) {
    const int pix = tile * PX + u * 16 + col;
    const int y = pix / W;
    const int xo = pix - y * W;
    if (full) {
#pragma unroll
      for (int r = 0; r < 8; ++r) {
        const int m = mtile * 16 + halfo + r;
        float v = acc[u][r] + bias[m];
        if (act) v = lrelu_f(v);
        out[(((size_t)b * Cout + m) * H + y) * W + xo] = v;
      }
    } else {
#pragma unroll
      for (int r = 0; r < 8; ++r) {
        const int m = mtile * 16 + halfo + r;
        if (m < Cout) {
          float v = acc[u][r] + bias[m];
          if (act) v = lrelu_f(v);
          out[(((size_t)b * Cout + m) * H + y) * W + xo] = v;
        }
      }
    }
  }
}

// ---------------------------------------------------------------------------
// Modulated deformable conv (DCNv2): bilinear gather + modulation staged to
// LDS as the B-panel of a (64 x 576 x 32px) WMMA GEMM.
// om = raw conv output, 216 ch: [0:72)=py_off, [72:144)=px_off, [144:216)=mask
// grid = (H*W/32, B); block = 128 = 32 pixels x 4 group-lanes for staging
// (each thread does 2 deform groups), 4 waves x (16 cout x 32 px) GEMM.
// ---------------------------------------------------------------------------
__global__ __launch_bounds__(128)
void dcn_wmma(const float* __restrict__ x, const float* __restrict__ om,
              const __bf16* __restrict__ wpack, const float* __restrict__ bias,
              float* __restrict__ out, int H, int W, int act) {
  __shared__ __attribute__((aligned(16))) __bf16 ldsB[32 * DCNK];
  const int b = blockIdx.y;
  const int tile = blockIdx.x;
  const int tid = threadIdx.x;
  const size_t hw = (size_t)H * W;

  // ---- stage: bilinear-sample 8 channels of each deform group at 9 taps
  {
    const int p = tid & 31;
    const int tc = tid >> 5;                 // 0..3 -> groups {tc, tc+4}
    const int pix = tile * 32 + p;
    const int y = pix / W;
    const int x0 = pix - y * W;
    const float* omB = om + (size_t)b * OMCH * hw;
    const size_t po = (size_t)y * W + x0;
#pragma unroll
    for (int gi = 0; gi < 2; ++gi) {
      const int g = tc + gi * 4;
      const float* xg = x + ((size_t)b * CH + g * 8) * hw;
#pragma unroll
      for (int k = 0; k < 9; ++k) {
        const int ky = k / 3 - 1, kx = k % 3 - 1;
        const float pyo = omB[(size_t)(g * 9 + k) * hw + po];
        const float pxo = omB[(size_t)(72 + g * 9 + k) * hw + po];
        const float mz  = omB[(size_t)(144 + g * 9 + k) * hw + po];
        const float mask = 1.f / (1.f + __expf(-mz));
        const float py = (float)(y + ky) + pyo;
        const float px = (float)(x0 + kx) + pxo;
        const float yf = floorf(py), xf = floorf(px);
        const float wy = py - yf, wx = px - xf;
        const int iy0 = (int)yf, ix0 = (int)xf;
        const int iy1 = iy0 + 1, ix1 = ix0 + 1;
        const float vy0 = (iy0 >= 0 && iy0 < H) ? 1.f : 0.f;
        const float vy1 = (iy1 >= 0 && iy1 < H) ? 1.f : 0.f;
        const float vx0 = (ix0 >= 0 && ix0 < W) ? 1.f : 0.f;
        const float vx1 = (ix1 >= 0 && ix1 < W) ? 1.f : 0.f;
        const int cy0 = min(max(iy0, 0), H - 1), cy1 = min(max(iy1, 0), H - 1);
        const int cx0 = min(max(ix0, 0), W - 1), cx1 = min(max(ix1, 0), W - 1);
        const float w00 = (1.f - wy) * (1.f - wx) * vy0 * vx0 * mask;
        const float w01 = (1.f - wy) * wx * vy0 * vx1 * mask;
        const float w10 = wy * (1.f - wx) * vy1 * vx0 * mask;
        const float w11 = wy * wx * vy1 * vx1 * mask;
        const int o00 = cy0 * W + cx0, o01 = cy0 * W + cx1;
        const int o10 = cy1 * W + cx0, o11 = cy1 * W + cx1;
#pragma unroll
        for (int c = 0; c < 8; ++c) {
          const float* chan = xg + (size_t)c * hw;
          float v = w00 * chan[o00] + w01 * chan[o01] +
                    w10 * chan[o10] + w11 * chan[o11];
          ldsB[p * DCNK + (g * 8 + c) * 9 + k] = (__bf16)v;
        }
      }
    }
  }
  __syncthreads();

  // ---- GEMM: out[64, px32] = w[64,576] * val[576, px32]
  const int wave = tid >> 5;
  const int lane = tid & 31;
  const int col = lane & 15;
  const int halfo = (lane >= 16) ? 8 : 0;
  const __bf16* arow = wpack + ((size_t)(wave * 16 + col)) * DCNK + halfo;
  v8f acc0 = {}, acc1 = {};
  wmma_panel2(arow, ldsB + col * DCNK + halfo,
              ldsB + (16 + col) * DCNK + halfo, DCNK, acc0, acc1);

#pragma unroll
  for (int u = 0; u < 2; ++u) {
    const v8f& acc = u ? acc1 : acc0;
    const int pix = tile * 32 + u * 16 + col;
    const int y = pix / W;
    const int xo = pix - y * W;
#pragma unroll
    for (int r = 0; r < 8; ++r) {
      const int m = wave * 16 + halfo + r;
      float v = acc[r] + bias[m];
      if (act) v = lrelu_f(v);
      out[(((size_t)b * CH + m) * H + y) * W + xo] = v;
    }
  }
}

// ---------------------------------------------------------------------------
// Bilinear 2x upsample, grid convention c = (o+0.5)/2 - 0.5 with edge clamp.
// ---------------------------------------------------------------------------
__global__ void up2x_kernel(const float* __restrict__ in, float* __restrict__ out,
                            int C, int H, int W, int Bn) {
  const int Ho = 2 * H, Wo = 2 * W;
  const size_t total = (size_t)Bn * C * Ho * Wo;
  for (size_t i = (size_t)blockIdx.x * 256 + threadIdx.x; i < total;
       i += (size_t)gridDim.x * 256) {
    const int xo = (int)(i % Wo);
    size_t r = i / Wo;
    const int yo = (int)(r % Ho); r /= Ho;
    const int c = (int)(r % C);
    const int b = (int)(r / C);
    const float cy = (yo + 0.5f) * 0.5f - 0.5f;
    const float cx = (xo + 0.5f) * 0.5f - 0.5f;
    const float fy = floorf(cy), fx = floorf(cx);
    const float ty = cy - fy, tx = cx - fx;
    const int y0 = min(max((int)fy, 0), H - 1);
    const int y1 = min(max((int)fy + 1, 0), H - 1);
    const int x0 = min(max((int)fx, 0), W - 1);
    const int x1 = min(max((int)fx + 1, 0), W - 1);
    const float* p = in + ((size_t)b * C + c) * H * W;
    const float v0 = p[y0 * W + x0] * (1.f - tx) + p[y0 * W + x1] * tx;
    const float v1 = p[y1 * W + x0] * (1.f - tx) + p[y1 * W + x1] * tx;
    out[i] = v0 * (1.f - ty) + v1 * ty;
  }
}

// ---------------------------------------------------------------------------
// Host orchestration
// ---------------------------------------------------------------------------
extern "C" void kernel_launch(void* const* d_in, const int* in_sizes, int n_in,
                              void* d_out, int out_size, void* d_ws, size_t ws_size,
                              hipStream_t stream) {
  (void)in_sizes; (void)n_in; (void)out_size; (void)ws_size;
  const int Bn = 2;
  const int H1 = 128, W1 = 128, H2 = 64, W2 = 64, H3 = 32, W3 = 32;

  const float* x1 = (const float*)d_in[0];
  const float* x2 = (const float*)d_in[1];
  const float* x3 = (const float*)d_in[2];
  // params flattened in dict insertion order
  const float* L3_oc2_w = (const float*)d_in[3];  const float* L3_oc2_b = (const float*)d_in[4];
  const float* L3_ow = (const float*)d_in[5];     const float* L3_ob = (const float*)d_in[6];
  const float* L3_w  = (const float*)d_in[7];     const float* L3_b  = (const float*)d_in[8];
  const float* L2_oc2_w = (const float*)d_in[9];  const float* L2_oc2_b = (const float*)d_in[10];
  const float* L2_oc3_w = (const float*)d_in[11]; const float* L2_oc3_b = (const float*)d_in[12];
  const float* L2_ow = (const float*)d_in[13];    const float* L2_ob = (const float*)d_in[14];
  const float* L2_w  = (const float*)d_in[15];    const float* L2_b  = (const float*)d_in[16];
  const float* L2_fc_w = (const float*)d_in[17];  const float* L2_fc_b = (const float*)d_in[18];
  const float* L1_oc2_w = (const float*)d_in[19]; const float* L1_oc2_b = (const float*)d_in[20];
  const float* L1_oc3_w = (const float*)d_in[21]; const float* L1_oc3_b = (const float*)d_in[22];
  const float* L1_ow = (const float*)d_in[23];    const float* L1_ob = (const float*)d_in[24];
  const float* L1_w  = (const float*)d_in[25];    const float* L1_b  = (const float*)d_in[26];
  const float* L1_fc_w = (const float*)d_in[27];  const float* L1_fc_b = (const float*)d_in[28];
  const float* cas_oc2_w = (const float*)d_in[29]; const float* cas_oc2_b = (const float*)d_in[30];
  const float* cas_ow = (const float*)d_in[31];   const float* cas_ob = (const float*)d_in[32];
  const float* cas_w  = (const float*)d_in[33];   const float* cas_b  = (const float*)d_in[34];

  char* base = (char*)d_ws;
  size_t off = 0;
  auto carve = [&](size_t bytes) -> void* {
    void* p = base + off;
    off = (off + bytes + 255) & ~(size_t)255;
    return p;
  };
  const size_t P3 = (size_t)Bn * CH * H3 * W3 * 4;
  const size_t P2 = (size_t)Bn * CH * H2 * W2 * 4;
  const size_t P1 = (size_t)Bn * CH * H1 * W1 * 4;
  float* A3 = (float*)carve(P3);
  float* B3 = (float*)carve(P3);
  float* C2 = (float*)carve(P2);
  float* D2 = (float*)carve(P2);
  float* E2 = (float*)carve(P2);
  float* U2 = (float*)carve(P2);
  float* F1 = (float*)carve(P1);
  float* G1 = (float*)carve(P1);
  float* Hb = (float*)carve(P1);
  float* U1 = (float*)carve(P1);
  float* OM = (float*)carve((size_t)Bn * OMCH * H1 * W1 * 4);

  // bf16 weight packs: {source, Cout, CoutPad, Cin}
  struct PackDesc { const float* w; int Cout; int CoutPad; int Cin; __bf16* buf; };
  PackDesc pk[16] = {
      {L3_oc2_w, 64, 64, 64, nullptr},  {L3_ow, 216, 224, 64, nullptr},
      {L3_w, 64, 64, 64, nullptr},      {L2_oc2_w, 64, 64, 128, nullptr},
      {L2_oc3_w, 64, 64, 64, nullptr},  {L2_ow, 216, 224, 64, nullptr},
      {L2_w, 64, 64, 64, nullptr},      {L2_fc_w, 64, 64, 128, nullptr},
      {L1_oc2_w, 64, 64, 128, nullptr}, {L1_oc3_w, 64, 64, 64, nullptr},
      {L1_ow, 216, 224, 64, nullptr},   {L1_w, 64, 64, 64, nullptr},
      {L1_fc_w, 64, 64, 128, nullptr},  {cas_oc2_w, 64, 64, 64, nullptr},
      {cas_ow, 216, 224, 64, nullptr},  {cas_w, 64, 64, 64, nullptr}};
  for (int i = 0; i < 16; ++i) {
    const int Kp = pk[i].Cin * 9;
    pk[i].buf = (__bf16*)carve((size_t)pk[i].CoutPad * Kp * 2);
    const int total = pk[i].CoutPad * Kp;
    pack_w_kernel<<<dim3((total + 255) / 256), dim3(256), 0, stream>>>(
        pk[i].w, pk[i].buf, pk[i].Cout, pk[i].CoutPad, Kp);
  }

  auto conv = [&](const float* in0, int C0, float s0, const float* in1, int C1,
                  float s1, const __bf16* wpk, const float* bias, float* out,
                  int Cout, int Mtiles, int H, int W, int act) {
    if (C0 + C1 > 64) {   // Kp = 1152: 16-pixel panel
      dim3 g((H * W) / 16, Bn, (Mtiles + 3) / 4);
      conv3x3_wmma<1><<<g, dim3(128), 0, stream>>>(in0, C0, s0, in1, C1, s1,
                                                   wpk, bias, out, Cout, Mtiles,
                                                   H, W, act);
    } else {              // Kp = 576: 32-pixel panel, 2 WMMAs per A-fragment
      dim3 g((H * W) / 32, Bn, (Mtiles + 3) / 4);
      conv3x3_wmma<2><<<g, dim3(128), 0, stream>>>(in0, C0, s0, in1, C1, s1,
                                                   wpk, bias, out, Cout, Mtiles,
                                                   H, W, act);
    }
  };
  auto dcn = [&](const float* x, const float* omb, const __bf16* wpk,
                 const float* bias, float* out, int H, int W, int act) {
    dcn_wmma<<<dim3((H * W) / 32, Bn), dim3(128), 0, stream>>>(x, omb, wpk,
                                                               bias, out, H, W, act);
  };
  auto up = [&](const float* in, float* out, int H, int W) {
    const size_t total = (size_t)Bn * CH * 4 * H * W;
    int blocks = (int)((total + 255) / 256);
    if (blocks > 65535) blocks = 65535;
    up2x_kernel<<<dim3(blocks), dim3(256), 0, stream>>>(in, out, CH, H, W, Bn);
  };

  // ---- L3 ----
  conv(x3, 64, 1.f, nullptr, 0, 0.f, pk[0].buf, L3_oc2_b, A3, 64, 4, H3, W3, 1);   // L3_off
  conv(A3, 64, 1.f, nullptr, 0, 0.f, pk[1].buf, L3_ob, OM, 216, 14, H3, W3, 0);    // om3
  dcn(x3, OM, pk[2].buf, L3_b, B3, H3, W3, 1);                                     // L3_fea
  // ---- L2 ----
  up(A3, U2, H3, W3);                                                              // L3_off up
  conv(x2, 64, 1.f, U2, 64, 2.f, pk[3].buf, L2_oc2_b, C2, 64, 4, H2, W2, 1);       // L2_off (1)
  conv(C2, 64, 1.f, nullptr, 0, 0.f, pk[4].buf, L2_oc3_b, D2, 64, 4, H2, W2, 1);   // L2_off (2)
  conv(D2, 64, 1.f, nullptr, 0, 0.f, pk[5].buf, L2_ob, OM, 216, 14, H2, W2, 0);    // om2
  dcn(x2, OM, pk[6].buf, L2_b, C2, H2, W2, 0);                                     // L2_fea raw
  up(B3, U2, H3, W3);                                                              // L3_fea up
  conv(C2, 64, 1.f, U2, 64, 1.f, pk[7].buf, L2_fc_b, E2, 64, 4, H2, W2, 1);        // L2_fea
  // ---- L1 ----
  up(D2, U1, H2, W2);                                                              // L2_off up
  conv(x1, 64, 1.f, U1, 64, 2.f, pk[8].buf, L1_oc2_b, F1, 64, 4, H1, W1, 1);       // L1_off (1)
  conv(F1, 64, 1.f, nullptr, 0, 0.f, pk[9].buf, L1_oc3_b, G1, 64, 4, H1, W1, 1);   // L1_off (2)
  conv(G1, 64, 1.f, nullptr, 0, 0.f, pk[10].buf, L1_ob, OM, 216, 14, H1, W1, 0);   // om1
  dcn(x1, OM, pk[11].buf, L1_b, F1, H1, W1, 0);                                    // L1_fea raw
  up(E2, U1, H2, W2);                                                              // L2_fea up
  conv(F1, 64, 1.f, U1, 64, 1.f, pk[12].buf, L1_fc_b, Hb, 64, 4, H1, W1, 0);       // L1_fea
  // ---- cascade ----
  conv(Hb, 64, 1.f, nullptr, 0, 0.f, pk[13].buf, cas_oc2_b, G1, 64, 4, H1, W1, 1); // cas off
  conv(G1, 64, 1.f, nullptr, 0, 0.f, pk[14].buf, cas_ob, OM, 216, 14, H1, W1, 0);  // om cas
  dcn(Hb, OM, pk[15].buf, cas_b, (float*)d_out, H1, W1, 1);                        // output
}
